// STFTTransferFunction_15779709845962
// MI455X (gfx1250) — matrix operand order repaired and checked by reading, not compiled, not run on hardware
//
#include <hip/hip_runtime.h>
#include <math.h>

// ---------------------------------------------------------------------------
// STFT transfer-function synthesis as a fused WMMA GEMM on gfx1250 (MI455X).
//
//   frames[b,f,n] = sum_k  Re(z_k^(f+1)) * C[k,n] + Im(z_k^(f+1)) * S[k,n]
//   out[b,t]      = frames[b, t/256, t%256] + frames[b, t/256-1, t%256+256]
//
// GEMM: M = B*F = 131072, K = 2*257 -> 544 padded, N = 512  (~69 GFLOP).
// Per-WG: A panel (128x544 f16 power chains) + basis slice (128 cols) in LDS
// (279.5 KB < 320 KB/WGP), v_wmma_f32_16x16x32_f16 inner loop,
// atomicAdd overlap-add epilogue; trim guard only on the single (mi,v)
// unroll iteration that can ever go out of range (f==127 tiles).
// ---------------------------------------------------------------------------

typedef __attribute__((ext_vector_type(16))) _Float16 v16h;
typedef __attribute__((ext_vector_type(8)))  float    v8f;
typedef __attribute__((ext_vector_type(2)))  _Float16 h2;

#define NCOEF 257
#define NFRM  128
#define WIN   512
#define STEP  256
#define NSAMP 32768
#define KPAD  544          // 2*257 = 514 padded up to multiple of 32
#define LDK   546          // panel row stride in halves (273 dwords, odd -> conflict-friendly)
#define NBLK  4            // column blocks per batch row
#define NCOLS 128          // output columns per block (NBLK*NCOLS = 512)

// 1/sqrt(512) and 2/sqrt(512)
#define W_EDGE 0.044194173824159216f
#define W_MID  0.08838834764831843f

// ---------------------------------------------------------------------------
// Kernel 1: precompute f16 irfft basis, layout bg[n][krow], stride LDK halves.
//   krow = 2k   : w_k * cos(2*pi*k*n/512)
//   krow = 2k+1 : (k==0 || k==256) ? 0 : -w_k * sin(2*pi*k*n/512)
// Rows >= 514 are zero padding for the K dimension.
// ---------------------------------------------------------------------------
__global__ void stft_basis_kernel(_Float16* __restrict__ bg) {
    int i = blockIdx.x * blockDim.x + threadIdx.x;   // over WIN*LDK entries
    if (i >= WIN * LDK) return;
    int n    = i / LDK;
    int krow = i % LDK;
    float v = 0.0f;
    if (krow < 2 * NCOEF) {
        int k = krow >> 1;
        float ang = (float)(k * n) * 0.012271846303085129f;  // pi/256
        float s, c;
        sincosf(ang, &s, &c);
        bool edge = (k == 0) || (k == NCOEF - 1);
        float w = edge ? W_EDGE : W_MID;
        v = (krow & 1) ? (edge ? 0.0f : -w * s) : (w * c);
    }
    bg[i] = (_Float16)v;
}

// ---------------------------------------------------------------------------
// Kernel 2: zero the output (harness poisons it; overlap-add uses atomics).
// ---------------------------------------------------------------------------
__global__ void stft_zero_kernel(float4* __restrict__ p, int n4) {
    int i = blockIdx.x * blockDim.x + threadIdx.x;
    if (i < n4) p[i] = make_float4(0.f, 0.f, 0.f, 0.f);
}

// ---------------------------------------------------------------------------
// Kernel 3: fused power-chain + WMMA GEMM + overlap-add.
// grid = (NBLK, batch), block = 256 threads (8 wave32).
// dynamic LDS: A panel [NFRM][LDK] halves + B panel [NCOLS][LDK] halves.
// ---------------------------------------------------------------------------
__global__ __launch_bounds__(256)
void stft_tf_kernel(const float* __restrict__ tf,
                    const _Float16* __restrict__ bg,
                    float* __restrict__ out) {
    extern __shared__ _Float16 smem[];
    _Float16* Ap = smem;                       // [NFRM][LDK]
    _Float16* Bp = smem + NFRM * LDK;          // [NCOLS][LDK] (column-major basis)

    const int b    = blockIdx.y;
    const int nb   = blockIdx.x;
    const int tid  = threadIdx.x;
    const int lane = tid & 31;
    const int w    = tid >> 5;

    // ---- Phase 0: copy basis slice (columns nb*NCOLS .. +NCOLS) to LDS ----
    {
        const uint32_t* src = (const uint32_t*)(bg + (size_t)nb * NCOLS * LDK);
        uint32_t*       dst = (uint32_t*)Bp;
        const int n32 = NCOLS * LDK / 2;       // 34944 dwords, coalesced
        for (int i = tid; i < n32; i += 256) dst[i] = src[i];
    }

    // ---- Phase 1: complex power chains -> A panel (f16, interleaved re/im) --
    for (int k = tid; k < KPAD / 2; k += 256) {            // k in [0, 272)
        if (k < NCOEF) {
            float x = tf[(size_t)b * (2 * NCOEF) + k];
            float y = tf[(size_t)b * (2 * NCOEF) + NCOEF + k];
            float r  = fminf(fmaxf(x, 0.0f), 1.0f) * 0.9999f;
            float th = fminf(fmaxf(y, -1.0f), 1.0f) * 3.14159265358979323846f;
            float s, c;
            sincosf(th, &s, &c);
            float zr = r * c;          // real2 = real*cos(imag)
            float zi = zr * s;         // imag2 = real2*sin(imag) (faithful)
            float cr = zr, ci = zi;    // cumprod starts at z^1
            for (int f = 0; f < NFRM; ++f) {
                h2 p; p[0] = (_Float16)cr; p[1] = (_Float16)ci;
                *(h2*)(Ap + f * LDK + 2 * k) = p;
                float t0 = cr * zr - ci * zi;
                float t1 = cr * zi + ci * zr;
                cr = t0; ci = t1;
            }
        } else {  // K padding columns 514..543 -> zeros
            h2 p; p[0] = (_Float16)0.f; p[1] = (_Float16)0.f;
            for (int f = 0; f < NFRM; ++f)
                *(h2*)(Ap + f * LDK + 2 * k) = p;
        }
    }
    __syncthreads();

    // ---- Phase 2: WMMA GEMM  C[128 x NCOLS] = A[128 x KPAD] * B[KPAD x NCOLS]
    // 8 waves: (w&3) picks 32 rows (2 M-tiles), (w>>2) picks 64 cols (4 N-tiles)
    const int mbase = (w & 3) * 32;
    const int nbase = (w >> 2) * 64;

    v8f acc[2][4];
    #pragma unroll
    for (int mi = 0; mi < 2; ++mi)
        #pragma unroll
        for (int ni = 0; ni < 4; ++ni) {
            v8f z = {0.f, 0.f, 0.f, 0.f, 0.f, 0.f, 0.f, 0.f};
            acc[mi][ni] = z;
        }

    const int mrow0 = mbase + (lane & 15);
    const int ncol0 = nbase + (lane & 15);
    const int hsel  = lane >> 4;

    for (int kk = 0; kk < KPAD; kk += 32) {
        // A fragments: ISA 16-bit A 16x32 layout -> lane (m, h): K = 8h + e + (e&8)
        v16h afrag[2];
        #pragma unroll
        for (int mi = 0; mi < 2; ++mi) {
            const _Float16* row = Ap + (mrow0 + mi * 16) * LDK + kk + 8 * hsel;
            #pragma unroll
            for (int e = 0; e < 16; ++e)
                afrag[mi][e] = row[e + (e & 8)];
        }
        // B fragments: lane (n, h): K = 8h + e + (e&8)  (basis stored [n][k] in LDS)
        v16h bfrag[4];
        #pragma unroll
        for (int ni = 0; ni < 4; ++ni) {
            const _Float16* col = Bp + (ncol0 + ni * 16) * LDK + kk + 8 * hsel;
            #pragma unroll
            for (int e = 0; e < 16; ++e)
                bfrag[ni][e] = col[e + (e & 8)];
        }
        #pragma unroll
        for (int mi = 0; mi < 2; ++mi)
            #pragma unroll
            for (int ni = 0; ni < 4; ++ni)
                acc[mi][ni] = __builtin_amdgcn_wmma_f32_16x16x32_f16(
                    false, afrag[mi], false, bfrag[ni],
                    (short)0, acc[mi][ni], false, false);
    }

    // ---- Phase 3: overlap-add epilogue (t = f*STEP + n, trimmed) ----------
    // C/D layout: VGPR v, lane group g: m = v + 8g, n = lane&15.
    // Range proof: f = mbase + mi*16 + v + 8*lg <= 96+16+7+8 = 127, and
    // t = f*256 + ncol >= NSAMP requires f == 127, which after unrolling can
    // only occur in the (mi==1, v==7) iteration.  All other 60 atomics per
    // lane are provably in range (f <= 126 -> t <= 126*256+511 = 32767) and
    // are emitted unconditionally, with constant per-iteration offsets.
    const int lg   = lane >> 4;
    const int nloc = lane & 15;
    float* outb = out + (size_t)b * NSAMP;
    const int base = (mbase + 8 * lg) * STEP + nb * NCOLS + nbase + nloc;
    #pragma unroll
    for (int mi = 0; mi < 2; ++mi) {
        #pragma unroll
        for (int ni = 0; ni < 4; ++ni) {
            #pragma unroll
            for (int v = 0; v < 8; ++v) {
                const int t = base + (mi * 16 + v) * STEP + ni * 16;
                if (mi == 1 && v == 7) {          // compile-time: only f==127 candidates
                    if (t < NSAMP) atomicAdd(outb + t, acc[mi][ni][v]);
                } else {
                    atomicAdd(outb + t, acc[mi][ni][v]);
                }
            }
        }
    }
}

// ---------------------------------------------------------------------------
extern "C" void kernel_launch(void* const* d_in, const int* in_sizes, int n_in,
                              void* d_out, int out_size, void* d_ws, size_t ws_size,
                              hipStream_t stream) {
    const float* tf  = (const float*)d_in[0];
    float*       out = (float*)d_out;
    const int row_elems = 2 * NCOEF;                  // 514 per batch row
    const int batch = in_sizes[0] / row_elems;        // 1024

    _Float16* bg = (_Float16*)d_ws;                   // 512*546*2 = 559,104 B

    // 1) basis table (f16, [n][krow] stride LDK)
    {
        const int n = WIN * LDK;
        stft_basis_kernel<<<(n + 255) / 256, 256, 0, stream>>>(bg);
    }
    // 2) zero the output (atomic overlap-add target)
    {
        const int n4 = out_size / 4;
        stft_zero_kernel<<<(n4 + 255) / 256, 256, 0, stream>>>((float4*)d_out, n4);
    }
    // 3) fused power-chain + WMMA GEMM + overlap-add
    {
        dim3 grid(NBLK, batch);
        const size_t smem_bytes = (size_t)(NFRM + NCOLS) * LDK * sizeof(_Float16);
        stft_tf_kernel<<<grid, 256, smem_bytes, stream>>>(tf, bg, out);
    }
}
